// Attention_22428319220644
// MI455X (gfx1250) — matrix operand order repaired and checked by reading, not compiled, not run on hardware
//
#include <hip/hip_runtime.h>
#include <cstddef>

// ---------------------------------------------------------------------------
// CDNA5 / gfx1250 fused causal multi-head attention block.
// Wave32, v_wmma_f32_16x16x32_f16 everywhere; f32 accumulate, f16 operands.
// Register-resident double-buffered fragments, global_prefetch warming.
// Workspace usage: 40 MB (see kernel_launch).
// ---------------------------------------------------------------------------

typedef __attribute__((ext_vector_type(16))) _Float16 v16h;
typedef __attribute__((ext_vector_type(8)))  _Float16 v8h;
typedef __attribute__((ext_vector_type(8)))  float    v8f;

union AFrag { v16h v; v8h h2[2]; _Float16 e[16]; };
union CFrag { v8f  v; float e[8]; };

__device__ __forceinline__ v8f wmma_f16(v16h a, v16h b, v8f c) {
  // D = A(16x32,f16) * B(32x16,f16) + C(16x16,f32)
  return __builtin_amdgcn_wmma_f32_16x16x32_f16(
      /*neg_a=*/false, a, /*neg_b=*/false, b,
      /*c_mod=*/(short)0, c, /*reuse_a=*/false, /*reuse_b=*/false);
}

#define D_MODEL 1024
#define N_HEADS 16
#define D_HEAD  64
#define SEQ     2048
#define BATCH   2
#define NROWS   (BATCH * SEQ)          // 4096 tokens

// ---------------------------------------------------------------------------
// 1) LayerNorm: fp32 resid -> f16 normalized activations
// ---------------------------------------------------------------------------
__global__ __launch_bounds__(256) void ln_kernel(
    const float* __restrict__ x, const float* __restrict__ w,
    const float* __restrict__ bias, _Float16* __restrict__ xh) {
  const int row = blockIdx.x;                 // 0..4095
  const float* xr = x + (size_t)row * D_MODEL;
  float s = 0.f, ss = 0.f, vals[4];
#pragma unroll
  for (int i = 0; i < 4; ++i) {
    float v = xr[threadIdx.x + i * 256];
    vals[i] = v; s += v; ss += v * v;
  }
#pragma unroll
  for (int m = 1; m < 32; m <<= 1) {
    s  += __shfl_xor(s,  m, 32);
    ss += __shfl_xor(ss, m, 32);
  }
  __shared__ float red0[8], red1[8];
  const int wv = threadIdx.x >> 5;
  if ((threadIdx.x & 31) == 0) { red0[wv] = s; red1[wv] = ss; }
  __syncthreads();
  s = 0.f; ss = 0.f;
#pragma unroll
  for (int i = 0; i < 8; ++i) { s += red0[i]; ss += red1[i]; }
  const float mean = s * (1.0f / D_MODEL);
  const float var  = ss * (1.0f / D_MODEL) - mean * mean;
  const float rstd = rsqrtf(var + 1e-5f);
#pragma unroll
  for (int i = 0; i < 4; ++i) {
    const int c = threadIdx.x + i * 256;
    xh[(size_t)row * D_MODEL + c] =
        (_Float16)((vals[i] - mean) * rstd * w[c] + bias[c]);
  }
}

// ---------------------------------------------------------------------------
// 2) Convert+transpose weights to f16 B-fragment-friendly layouts:
//    Wq/k/v [h][d][e] -> Wt [h][e][d]   (lane=e, contiguous K=d)
//    Wo     [h][e][d] -> Wot [d][h*64+e] (lane=d, contiguous K=h*64+e)
// ---------------------------------------------------------------------------
__global__ __launch_bounds__(256) void wcvt_kernel(
    const float* __restrict__ Wq, const float* __restrict__ Wk,
    const float* __restrict__ Wv, const float* __restrict__ Wo,
    _Float16* __restrict__ Wqt, _Float16* __restrict__ Wkt,
    _Float16* __restrict__ Wvt, _Float16* __restrict__ Wot) {
  const int idx = blockIdx.x * 256 + threadIdx.x;   // 0 .. 4*2^20-1
  const int sel = idx >> 20;
  const int r   = idx & 1048575;
  if (sel < 3) {
    const int h = r >> 16, rem = r & 65535, d = rem >> 6, e = rem & 63;
    const float* W = (sel == 0) ? Wq : (sel == 1) ? Wk : Wv;
    _Float16* Wt   = (sel == 0) ? Wqt : (sel == 1) ? Wkt : Wvt;
    Wt[(size_t)h * 65536 + (size_t)e * 1024 + d] = (_Float16)W[r];
  } else {
    const int h = r >> 16, rem = r & 65535, e = rem >> 10, d = rem & 1023;
    Wot[(size_t)d * 1024 + h * 64 + e] = (_Float16)Wo[r];
  }
}

// ---------------------------------------------------------------------------
// 3) QKV projection. One wave = 32 token rows x one head (64 cols), for all
//    of q,k,v simultaneously: 24 WMMAs per K-chunk with B fragments reused
//    across both M-tiles in registers. A fragments double-buffered.
//    Q,K stored [b,h,s,e]; V stored transposed [b,h,e,s] for PV fragments.
// ---------------------------------------------------------------------------
__global__ __launch_bounds__(256) void qkv_kernel(
    const _Float16* __restrict__ xh,
    const _Float16* __restrict__ Wqt, const _Float16* __restrict__ Wkt,
    const _Float16* __restrict__ Wvt,
    const float* __restrict__ bq, const float* __restrict__ bk,
    const float* __restrict__ bv,
    _Float16* __restrict__ Q, _Float16* __restrict__ K,
    _Float16* __restrict__ Vt) {
  const int wave = blockIdx.x * 8 + (threadIdx.x >> 5);   // 0..2047
  const int lane = threadIdx.x & 31;
  const int hf = lane >> 4, ll = lane & 15;
  const int h = wave & 15, rowTile = wave >> 4;           // 128 tiles of 32
  const int rowBase = rowTile * 32;

  v8f accq[2][4] = {}, acck[2][4] = {}, accv[2][4] = {};
  const _Float16* arow0 = xh + (size_t)(rowBase + ll) * D_MODEL;
  const _Float16* arow1 = arow0 + (size_t)16 * D_MODEL;
  const _Float16* wqh = Wqt + (size_t)h * 65536;
  const _Float16* wkh = Wkt + (size_t)h * 65536;
  const _Float16* wvh = Wvt + (size_t)h * 65536;

  AFrag aCur[2], aNxt[2];
  aCur[0].h2[0] = *(const v8h*)(arow0 + hf * 8);
  aCur[0].h2[1] = *(const v8h*)(arow0 + 16 + hf * 8);
  aCur[1].h2[0] = *(const v8h*)(arow1 + hf * 8);
  aCur[1].h2[1] = *(const v8h*)(arow1 + 16 + hf * 8);

  for (int kc = 0; kc < D_MODEL; kc += 32) {
    // All 12 B fragments in distinct registers -> loads clause together,
    // waits amortize instead of a full s_wait_loadcnt 0 per WMMA.
    AFrag bfq[4], bfk[4], bfv[4];
#pragma unroll
    for (int t = 0; t < 4; ++t) {
      const size_t boff = (size_t)(t * 16 + ll) * 1024 + kc + hf * 8;
      bfq[t].h2[0] = *(const v8h*)(wqh + boff);
      bfq[t].h2[1] = *(const v8h*)(wqh + boff + 16);
      bfk[t].h2[0] = *(const v8h*)(wkh + boff);
      bfk[t].h2[1] = *(const v8h*)(wkh + boff + 16);
      bfv[t].h2[0] = *(const v8h*)(wvh + boff);
      bfv[t].h2[1] = *(const v8h*)(wvh + boff + 16);
    }
    const int kn = kc + 32;
    if (kn < D_MODEL) {
      // double-buffer A for next chunk
      aNxt[0].h2[0] = *(const v8h*)(arow0 + kn + hf * 8);
      aNxt[0].h2[1] = *(const v8h*)(arow0 + kn + 16 + hf * 8);
      aNxt[1].h2[0] = *(const v8h*)(arow1 + kn + hf * 8);
      aNxt[1].h2[1] = *(const v8h*)(arow1 + kn + 16 + hf * 8);
      if (kn + 32 < D_MODEL) {
        // warm L2/WGP$ one chunk further ahead (global_prefetch_b8)
#pragma unroll
        for (int t = 0; t < 4; ++t) {
          const size_t poff = (size_t)(t * 16 + ll) * 1024 + kn + 32 + hf * 16;
          __builtin_prefetch(wqh + poff, 0, 1);
          __builtin_prefetch(wkh + poff, 0, 1);
          __builtin_prefetch(wvh + poff, 0, 1);
        }
      }
    }
#pragma unroll
    for (int t = 0; t < 4; ++t) {
#pragma unroll
      for (int mt = 0; mt < 2; ++mt) {
        accq[mt][t] = wmma_f16(aCur[mt].v, bfq[t].v, accq[mt][t]);
        acck[mt][t] = wmma_f16(aCur[mt].v, bfk[t].v, acck[mt][t]);
        accv[mt][t] = wmma_f16(aCur[mt].v, bfv[t].v, accv[mt][t]);
      }
    }
    aCur[0] = aNxt[0];
    aCur[1] = aNxt[1];
  }

  const int b = rowBase >> 11;            // row / SEQ
  const size_t qkBase = (size_t)(b * N_HEADS + h) * SEQ * D_HEAD;
  const size_t vtBase = (size_t)(b * N_HEADS + h) * D_HEAD * SEQ;
#pragma unroll
  for (int mt = 0; mt < 2; ++mt) {
    const int sBase = (rowBase & (SEQ - 1)) + mt * 16;
#pragma unroll
    for (int t = 0; t < 4; ++t) {
      const int e = t * 16 + ll;
      const float bqv = bq[h * 64 + e];
      const float bkv = bk[h * 64 + e];
      const float bvv = bv[h * 64 + e];
#pragma unroll
      for (int r = 0; r < 8; ++r) {
        const int s = sBase + r + hf * 8;
        Q[qkBase + (size_t)s * 64 + e] = (_Float16)(accq[mt][t][r] + bqv);
        K[qkBase + (size_t)s * 64 + e] = (_Float16)(acck[mt][t][r] + bkv);
        Vt[vtBase + (size_t)e * SEQ + s] = (_Float16)(accv[mt][t][r] + bvv);
      }
    }
  }
}

// ---------------------------------------------------------------------------
// 4) Causal flash attention. One wave (block of 32) per 16-query tile per
//    (b,h). Online softmax; P transposed C->A layout through padded LDS.
//    K and V fragments all issued up front so V latency hides under the
//    softmax VALU work; next chunk prefetched.
// ---------------------------------------------------------------------------
__global__ __launch_bounds__(32) void attn_kernel(
    const _Float16* __restrict__ Q, const _Float16* __restrict__ K,
    const _Float16* __restrict__ Vt, _Float16* __restrict__ zh) {
  __shared__ _Float16 lds[16 * 40];      // padded stride: 40 halves = 80B
  const int lane = threadIdx.x & 31;
  const int hf = lane >> 4, ll = lane & 15;
  const int qTile = blockIdx.x & 127;    // S/16 = 128
  const int bh = blockIdx.x >> 7;        // b*16 + h
  const int b = bh >> 4, h = bh & 15;
  const size_t base  = (size_t)bh * SEQ * D_HEAD;
  const size_t vbase = (size_t)bh * D_HEAD * SEQ;
  const int qBase = qTile * 16;

  // Q fragments (pre-scaled by d_head^-0.5 = 0.125, exact in f16)
  AFrag aq[2];
  const _Float16* qrow = Q + base + (size_t)(qBase + ll) * 64;
#pragma unroll
  for (int t = 0; t < 2; ++t) {
    aq[t].h2[0] = *(const v8h*)(qrow + t * 32 + hf * 16);
    aq[t].h2[1] = *(const v8h*)(qrow + t * 32 + hf * 16 + 8);
#pragma unroll
    for (int i = 0; i < 16; ++i) aq[t].e[i] *= (_Float16)0.125f;
  }

  v8f o[4] = {};
  float m[8], l[8];
#pragma unroll
  for (int r = 0; r < 8; ++r) { m[r] = -1e30f; l[r] = 0.f; }

  const int kEnd = qBase + 16;
  for (int kb = 0; kb < kEnd; kb += 32) {
    const _Float16* krow0 = K + base + (size_t)(kb + ll) * 64;
    const _Float16* krow1 = krow0 + 16 * 64;

    // ---- issue ALL K and V fragment loads up front (distinct regs) -----
    AFrag bk00, bk01, bk10, bk11, bv[4];
    bk00.h2[0] = *(const v8h*)(krow0 + hf * 16);
    bk00.h2[1] = *(const v8h*)(krow0 + hf * 16 + 8);
    bk01.h2[0] = *(const v8h*)(krow0 + 32 + hf * 16);
    bk01.h2[1] = *(const v8h*)(krow0 + 32 + hf * 16 + 8);
    bk10.h2[0] = *(const v8h*)(krow1 + hf * 16);
    bk10.h2[1] = *(const v8h*)(krow1 + hf * 16 + 8);
    bk11.h2[0] = *(const v8h*)(krow1 + 32 + hf * 16);
    bk11.h2[1] = *(const v8h*)(krow1 + 32 + hf * 16 + 8);
#pragma unroll
    for (int t = 0; t < 4; ++t) {
      const _Float16* vrow =
          Vt + vbase + (size_t)(t * 16 + ll) * SEQ + kb + hf * 16;
      bv[t].h2[0] = *(const v8h*)(vrow);
      bv[t].h2[1] = *(const v8h*)(vrow + 8);
    }
    if (kb + 32 < kEnd) {   // warm next chunk (rows stay in-bounds)
      __builtin_prefetch(krow0 + 32 * 64, 0, 1);
      __builtin_prefetch(krow1 + 32 * 64, 0, 1);
#pragma unroll
      for (int t = 0; t < 4; ++t)
        __builtin_prefetch(Vt + vbase + (size_t)(t * 16 + ll) * SEQ + kb + 32,
                           0, 1);
    }

    // ---- scores: two 16x16 tiles over 32 keys --------------------------
    v8f s0v = {}, s1v = {};
    s0v = wmma_f16(aq[0].v, bk00.v, s0v);
    s0v = wmma_f16(aq[1].v, bk01.v, s0v);
    s1v = wmma_f16(aq[0].v, bk10.v, s1v);
    s1v = wmma_f16(aq[1].v, bk11.v, s1v);

    CFrag c0, c1; c0.v = s0v; c1.v = s1v;

    // ---- causal mask + online softmax (V loads still in flight) --------
    float alpha[8];
#pragma unroll
    for (int r = 0; r < 8; ++r) {
      const int q = qBase + r + hf * 8;
      const int k0 = kb + ll, k1 = k0 + 16;
      if (k0 > q) c0.e[r] = -1e30f;
      if (k1 > q) c1.e[r] = -1e30f;
      float v = fmaxf(c0.e[r], c1.e[r]);
#pragma unroll
      for (int msk = 1; msk < 16; msk <<= 1)
        v = fmaxf(v, __shfl_xor(v, msk, 32));   // reduce within 16-lane half
      const float mn = fmaxf(m[r], v);
      alpha[r] = __expf(m[r] - mn);
      m[r] = mn;
      c0.e[r] = __expf(c0.e[r] - mn);
      c1.e[r] = __expf(c1.e[r] - mn);
      float rs = c0.e[r] + c1.e[r];
#pragma unroll
      for (int msk = 1; msk < 16; msk <<= 1) rs += __shfl_xor(rs, msk, 32);
      l[r] = l[r] * alpha[r] + rs;
    }
#pragma unroll
    for (int t = 0; t < 4; ++t)
#pragma unroll
      for (int r = 0; r < 8; ++r) o[t][r] *= alpha[r];

    // ---- P: C layout -> A layout via LDS (conflict-free, 16B aligned) --
#pragma unroll
    for (int r = 0; r < 8; ++r) {
      lds[(r + hf * 8) * 40 + ll]      = (_Float16)c0.e[r];
      lds[(r + hf * 8) * 40 + 16 + ll] = (_Float16)c1.e[r];
    }
    __syncthreads();
    AFrag p;
    p.h2[0] = *(const v8h*)(&lds[ll * 40 + hf * 16]);
    p.h2[1] = *(const v8h*)(&lds[ll * 40 + hf * 16 + 8]);
    __syncthreads();

    // ---- PV: o[t] += P(16x32) * V(32x16) ------------------------------
#pragma unroll
    for (int t = 0; t < 4; ++t) o[t] = wmma_f16(p.v, bv[t].v, o[t]);
  }

  // ---- epilogue: z[b,q, h*64+e] = O / l --------------------------------
  float rl[8];
#pragma unroll
  for (int r = 0; r < 8; ++r) rl[r] = 1.0f / l[r];
#pragma unroll
  for (int t = 0; t < 4; ++t) {
    const int col = h * 64 + t * 16 + ll;
#pragma unroll
    for (int r = 0; r < 8; ++r) {
      const int row = qBase + r + hf * 8;
      zh[((size_t)b * SEQ + row) * D_MODEL + col] = (_Float16)(o[t][r] * rl[r]);
    }
  }
}

// ---------------------------------------------------------------------------
// 5) Output projection: out[4096,1024] = z * Wo^T(+b_o), fp32 result.
//    32-row tiles, double-buffered A, distinct B fragment registers.
// ---------------------------------------------------------------------------
__global__ __launch_bounds__(256) void oproj_kernel(
    const _Float16* __restrict__ zh, const _Float16* __restrict__ Wot,
    const float* __restrict__ bo, float* __restrict__ out) {
  const int wave = blockIdx.x * 8 + (threadIdx.x >> 5);   // 0..2047
  const int lane = threadIdx.x & 31;
  const int hf = lane >> 4, ll = lane & 15;
  const int nGroup = wave & 15, rowTile = wave >> 4;      // 128 tiles of 32
  const int rowBase = rowTile * 32, nBase = nGroup * 64;

  v8f acc[2][4] = {};
  const _Float16* arow0 = zh + (size_t)(rowBase + ll) * D_MODEL;
  const _Float16* arow1 = arow0 + (size_t)16 * D_MODEL;

  AFrag aCur[2], aNxt[2];
  aCur[0].h2[0] = *(const v8h*)(arow0 + hf * 8);
  aCur[0].h2[1] = *(const v8h*)(arow0 + 16 + hf * 8);
  aCur[1].h2[0] = *(const v8h*)(arow1 + hf * 8);
  aCur[1].h2[1] = *(const v8h*)(arow1 + 16 + hf * 8);

  for (int kc = 0; kc < D_MODEL; kc += 32) {
    AFrag bb[4];
#pragma unroll
    for (int t = 0; t < 4; ++t) {
      const _Float16* brow =
          Wot + (size_t)(nBase + t * 16 + ll) * 1024 + kc + hf * 8;
      bb[t].h2[0] = *(const v8h*)(brow);
      bb[t].h2[1] = *(const v8h*)(brow + 16);
    }
    const int kn = kc + 32;
    if (kn < D_MODEL) {
      aNxt[0].h2[0] = *(const v8h*)(arow0 + kn + hf * 8);
      aNxt[0].h2[1] = *(const v8h*)(arow0 + kn + 16 + hf * 8);
      aNxt[1].h2[0] = *(const v8h*)(arow1 + kn + hf * 8);
      aNxt[1].h2[1] = *(const v8h*)(arow1 + kn + 16 + hf * 8);
      if (kn + 32 < D_MODEL) {
#pragma unroll
        for (int t = 0; t < 4; ++t)
          __builtin_prefetch(
              Wot + (size_t)(nBase + t * 16 + ll) * 1024 + kn + 32 + hf * 16,
              0, 1);
      }
    }
#pragma unroll
    for (int t = 0; t < 4; ++t) {
      acc[0][t] = wmma_f16(aCur[0].v, bb[t].v, acc[0][t]);
      acc[1][t] = wmma_f16(aCur[1].v, bb[t].v, acc[1][t]);
    }
    aCur[0] = aNxt[0];
    aCur[1] = aNxt[1];
  }
#pragma unroll
  for (int mt = 0; mt < 2; ++mt) {
#pragma unroll
    for (int t = 0; t < 4; ++t) {
      const int n = nBase + t * 16 + ll;
      const float bov = bo[n];
#pragma unroll
      for (int r = 0; r < 8; ++r)
        out[(size_t)(rowBase + mt * 16 + r + hf * 8) * D_MODEL + n] =
            acc[mt][t][r] + bov;
    }
  }
}

// ---------------------------------------------------------------------------
extern "C" void kernel_launch(void* const* d_in, const int* in_sizes, int n_in,
                              void* d_out, int out_size, void* d_ws,
                              size_t ws_size, hipStream_t stream) {
  (void)in_sizes; (void)n_in; (void)out_size; (void)ws_size;
  const float* resid = (const float*)d_in[0];
  // d_in[1] = attn_mask (unused by reference forward)
  const float* W_q = (const float*)d_in[2];
  const float* W_k = (const float*)d_in[3];
  const float* W_v = (const float*)d_in[4];
  const float* W_o = (const float*)d_in[5];
  const float* b_q = (const float*)d_in[6];
  const float* b_k = (const float*)d_in[7];
  const float* b_v = (const float*)d_in[8];
  const float* b_o = (const float*)d_in[9];
  const float* ln_w = (const float*)d_in[10];
  const float* ln_b = (const float*)d_in[11];
  float* out = (float*)d_out;

  // Workspace layout (40 MB total):
  char* w = (char*)d_ws;
  _Float16* xh  = (_Float16*)(w);                          //  8 MB (reused as z)
  _Float16* Wqt = (_Float16*)(w + ((size_t)8  << 20));     //  2 MB
  _Float16* Wkt = (_Float16*)(w + ((size_t)10 << 20));     //  2 MB
  _Float16* Wvt = (_Float16*)(w + ((size_t)12 << 20));     //  2 MB
  _Float16* Wot = (_Float16*)(w + ((size_t)14 << 20));     //  2 MB
  _Float16* Qb  = (_Float16*)(w + ((size_t)16 << 20));     //  8 MB
  _Float16* Kb  = (_Float16*)(w + ((size_t)24 << 20));     //  8 MB
  _Float16* Vtb = (_Float16*)(w + ((size_t)32 << 20));     //  8 MB
  _Float16* zh  = xh;  // xh dead after qkv_kernel; reuse for z

  ln_kernel<<<NROWS, 256, 0, stream>>>(resid, ln_w, ln_b, xh);
  wcvt_kernel<<<(4 * 1048576) / 256, 256, 0, stream>>>(
      W_q, W_k, W_v, W_o, Wqt, Wkt, Wvt, Wot);
  qkv_kernel<<<256, 256, 0, stream>>>(xh, Wqt, Wkt, Wvt, b_q, b_k, b_v,
                                      Qb, Kb, Vtb);
  attn_kernel<<<BATCH * N_HEADS * (SEQ / 16), 32, 0, stream>>>(Qb, Kb, Vtb, zh);
  oproj_kernel<<<256, 256, 0, stream>>>(zh, Wot, b_o, out);
}
